// MultiHead_54348516163892
// MI455X (gfx1250) — matrix operand order repaired and checked by reading, not compiled, not run on hardware
//
#include <hip/hip_runtime.h>
#include <cstddef>

// ---- problem constants ----
constexpr int Bsz = 4, Sq = 2048, Hd = 1024, NH = 16, DK = 64;
constexpr int Mtot = Bsz * Sq;   // 8192
constexpr int Ntot = 1024;       // NH*DK == Hd
constexpr int Ktot = 1024;

// ---- WMMA vector types (CDNA5 wave32) ----
typedef __attribute__((ext_vector_type(16))) __bf16          v16bf;
typedef __attribute__((ext_vector_type(8)))  float           v8f;
typedef __attribute__((ext_vector_type(8)))  unsigned short  v8us;
typedef __attribute__((ext_vector_type(16))) unsigned short  v16us;
// ---- TDM descriptor vector types ----
typedef __attribute__((ext_vector_type(4)))  unsigned int    u32x4;
typedef __attribute__((ext_vector_type(4)))  int             i32x4;
typedef __attribute__((ext_vector_type(8)))  int             i32x8;

static __device__ __forceinline__ unsigned short f2bf(float f) {
    unsigned int u = __builtin_bit_cast(unsigned int, f);
    unsigned int r = u + 0x7FFFu + ((u >> 16) & 1u);   // round-to-nearest-even
    return (unsigned short)(r >> 16);
}

static __device__ __forceinline__ v16bf mk16(v8us lo, v8us hi) {
    v16us u;
#pragma unroll
    for (int e = 0; e < 8; e++) { u[e] = lo[e]; u[8 + e] = hi[e]; }
    return __builtin_bit_cast(v16bf, u);
}

static __device__ __forceinline__ v8f zero8() {
    v8f z;
#pragma unroll
    for (int r = 0; r < 8; r++) z[r] = 0.0f;
    return z;
}

// =====================================================================
// Tensor Data Mover: async 2D bf16 tile copy, global -> LDS.
// Descriptor layout per CDNA5 ISA §8 (D# group0 128b + group1 256b).
// Issued wave-uniform (TDM ignores EXEC); caller waits on TENSORcnt.
// =====================================================================
static __device__ __forceinline__ void tdm_load_tile_bf16(
    const unsigned short* gbase,   // element ptr at tile origin
    unsigned ldsOff,               // LDS byte offset of destination
    unsigned tileRows, unsigned tileCols,
    unsigned rowStrideElems, unsigned tensorRows)
{
    unsigned long long ga = (unsigned long long)(size_t)gbase;  // byte address
    u32x4 g0;
    g0[0] = 1u;                                                  // count=1 (valid user D#)
    g0[1] = ldsOff;                                              // lds_addr [63:32]
    g0[2] = (unsigned)(ga & 0xFFFFFFFFu);                        // global_addr lo
    g0[3] = (unsigned)((ga >> 32) & 0x01FFFFFFu) | (2u << 30);   // global_addr[56:32], type=2

    i32x8 g1;
    g1[0] = (int)(1u << 16);                                     // data_size=1 (2B), wg_mask=0
    g1[1] = (int)((rowStrideElems & 0xFFFFu) << 16);             // tensor_dim0[15:0] @ [63:48]
    g1[2] = (int)(((rowStrideElems >> 16) & 0xFFFFu) |           // tensor_dim0[31:16]
                  ((tensorRows & 0xFFFFu) << 16));               // tensor_dim1[15:0]
    g1[3] = (int)(((tensorRows >> 16) & 0xFFFFu) |               // tensor_dim1[31:16]
                  ((tileCols & 0xFFFFu) << 16));                 // tile_dim0 @ [127:112]
    g1[4] = (int)(tileRows & 0xFFFFu);                           // tile_dim1, tile_dim2=0
    g1[5] = (int)rowStrideElems;                                 // tensor_dim0_stride[31:0]
    g1[6] = 0;                                                   // stride hi, dim1_stride lo
    g1[7] = 0;                                                   // dim1_stride hi

    i32x4 z4 = {0, 0, 0, 0};
#if __has_include(<hip/amd_detail/amd_gfx1250_TDM.h>)
    i32x8 z8 = {0, 0, 0, 0, 0, 0, 0, 0};
    __builtin_amdgcn_tensor_load_to_lds(g0, g1, z4, z4, z8, 0);  // 6-arg toolchain
#else
    __builtin_amdgcn_tensor_load_to_lds(g0, g1, z4, z4, 0);      // 5-arg ROCm 7.2
#endif
}

// =====================================================================
// Generic bf16-compute GEMM:  C[M,N] = A[M,K] * B[K,N] + bias[N]
//   TA/TB: float (converted to bf16 on stage-in) or unsigned short (bf16)
//   OUT_MODE: 0 = fp32 row-major, 1 = bf16 row-major,
//             2 = bf16 transposed attention-V layout [b,h,d,s]
// Block tile 128x256, 8 waves of 64x64, K-chunk 32.
// bf16 A operand is staged via the Tensor Data Mover (async tensor DMA).
// =====================================================================
constexpr int BM = 128, BN = 256, BK = 32, BSTR = 48;  // BSTR: padded LDS k-stride

template <typename TA, typename TB, int OUT_MODE>
__global__ __launch_bounds__(256) void gemm_bf16_wmma(
    const TA* __restrict__ A, const TB* __restrict__ Bm,
    const float* __restrict__ bias, void* __restrict__ Cout)
{
    __shared__ unsigned short As[BM * BK];    // [row][k]   8 KB
    __shared__ unsigned short Bs[BN * BSTR];  // [n][k]    24 KB (padded)

    const int t    = threadIdx.x;
    const int lane = t & 31;
    const int w    = t >> 5;
    const int m0 = blockIdx.x * BM, n0 = blockIdx.y * BN;
    const int wm = (w >> 2) * 64,   wn = (w & 3) * 64;
    const int lm = lane & 15,       lh = lane >> 4;

    v8f acc[4][4];
#pragma unroll
    for (int i = 0; i < 4; i++)
#pragma unroll
        for (int j = 0; j < 4; j++) acc[i][j] = zero8();

    // staging assignments
    const int ar  = t >> 1;          // A row 0..127
    const int ak  = (t & 1) * 16;    // A col half 0/16
    const int bk  = t & 31;          // B k row 0..31
    const int bnb = (t >> 5) * 32;   // B n base 0..224

    for (int kk = 0; kk < Ktot; kk += BK) {
        // ---- stage A tile ----
        if constexpr (sizeof(TA) == 2) {
            // bf16 source: async TDM tile copy, issued once per block by wave 0
            if (w == 0) {
                tdm_load_tile_bf16((const unsigned short*)A + (size_t)m0 * Ktot + kk,
                                   (unsigned)(size_t)(void*)As,
                                   BM, BK, (unsigned)Ktot, (unsigned)BM);
                __builtin_amdgcn_s_wait_tensorcnt(0);
            }
        } else {
            const TA* src = A + (size_t)(m0 + ar) * Ktot + kk + ak;
            unsigned short* dst = &As[ar * BK + ak];
#pragma unroll
            for (int v = 0; v < 4; v++) {
                float4 f = ((const float4*)src)[v];
                dst[v * 4 + 0] = f2bf(f.x); dst[v * 4 + 1] = f2bf(f.y);
                dst[v * 4 + 2] = f2bf(f.z); dst[v * 4 + 3] = f2bf(f.w);
            }
            if (kk + BK < Ktot)
                __builtin_prefetch(A + (size_t)(m0 + ar) * Ktot + kk + BK + ak, 0, 1);
        }
        // ---- stage B tile transposed: Bs[n][k] ----
        {
            const TB* src = Bm + (size_t)(kk + bk) * Ntot + n0 + bnb;
            if constexpr (sizeof(TB) == 4) {
#pragma unroll
                for (int v = 0; v < 8; v++) {
                    float4 f = ((const float4*)src)[v];
                    int nb = bnb + v * 4;
                    Bs[(nb + 0) * BSTR + bk] = f2bf(f.x);
                    Bs[(nb + 1) * BSTR + bk] = f2bf(f.y);
                    Bs[(nb + 2) * BSTR + bk] = f2bf(f.z);
                    Bs[(nb + 3) * BSTR + bk] = f2bf(f.w);
                }
            } else {
#pragma unroll
                for (int v = 0; v < 4; v++) {
                    uint4 u = ((const uint4*)src)[v];
                    unsigned int ws_[4] = {u.x, u.y, u.z, u.w};
#pragma unroll
                    for (int q = 0; q < 4; q++) {
                        int nb = bnb + v * 8 + q * 2;
                        Bs[(nb + 0) * BSTR + bk] = (unsigned short)(ws_[q] & 0xFFFFu);
                        Bs[(nb + 1) * BSTR + bk] = (unsigned short)(ws_[q] >> 16);
                    }
                }
            }
        }
        __syncthreads();

        // ---- fragments (ISA 7.12.2 layouts) ----
        v16bf af[4], bfv[4];
#pragma unroll
        for (int i = 0; i < 4; i++) {
            const unsigned short* p = &As[(wm + i * 16 + lm) * BK + lh * 8];
            af[i] = mk16(*(const v8us*)p, *(const v8us*)(p + 16));
        }
#pragma unroll
        for (int j = 0; j < 4; j++) {
            const unsigned short* p = &Bs[(wn + j * 16 + lm) * BSTR + lh * 8];
            bfv[j] = mk16(*(const v8us*)p, *(const v8us*)(p + 16));
        }
#pragma unroll
        for (int i = 0; i < 4; i++)
#pragma unroll
            for (int j = 0; j < 4; j++)
                acc[i][j] = __builtin_amdgcn_wmma_f32_16x16x32_bf16(
                    false, af[i], false, bfv[j], (short)0, acc[i][j], false, false);
        __syncthreads();
    }

    // ---- epilogue: bias add + store ----
#pragma unroll
    for (int i = 0; i < 4; i++) {
#pragma unroll
        for (int j = 0; j < 4; j++) {
            const int gn = n0 + wn + j * 16 + lm;
            const float bvv = bias[gn];
#pragma unroll
            for (int r = 0; r < 8; r++) {
                const int gm = m0 + wm + i * 16 + r + 8 * lh;
                const float val = acc[i][j][r] + bvv;
                if constexpr (OUT_MODE == 0) {
                    ((float*)Cout)[(size_t)gm * Ntot + gn] = val;
                } else if constexpr (OUT_MODE == 1) {
                    ((unsigned short*)Cout)[(size_t)gm * Ntot + gn] = f2bf(val);
                } else {
                    const int bb = gm >> 11, ss = gm & (Sq - 1);
                    const int hh = gn >> 6,  dd = gn & (DK - 1);
                    ((unsigned short*)Cout)[(((size_t)(bb * NH + hh)) * DK + dd) * Sq + ss] = f2bf(val);
                }
            }
        }
    }
}

// =====================================================================
// Flash attention: Q[b,s,h,d] bf16, K[b,s,h,d] bf16, Vt[b,h,d,s] bf16
// -> Ctx[b,s,h,d] bf16.  8 waves/block, each wave owns a 16-row Q tile.
// =====================================================================
constexpr int PSTR = 48;  // padded LDS stride for P reshuffle

__global__ __launch_bounds__(256) void attn_fwd_wmma(
    const unsigned short* __restrict__ Q, const unsigned short* __restrict__ Kw,
    const unsigned short* __restrict__ Vt, unsigned short* __restrict__ Ctx)
{
    __shared__ unsigned short Pl[8 * 16 * PSTR];  // per-wave 16x32 P buffer (12 KB)

    const int t = threadIdx.x;
    const int lane = t & 31, w = t >> 5;
    const int lm = lane & 15, lh = lane >> 4;

    const int bid = blockIdx.x;
    const int qb = bid & 15;
    const int h  = (bid >> 4) & 15;
    const int b  = bid >> 8;
    const int q0 = qb * 128 + w * 16;

    unsigned short* P = &Pl[w * 16 * PSTR];

    // Q A-fragments for d-chunks 0..31, 32..63
    v16bf qf[2];
    {
        const unsigned short* qrow = Q + ((size_t)(b * Sq + q0 + lm)) * Ntot + h * DK;
#pragma unroll
        for (int c = 0; c < 2; c++) {
            const unsigned short* p = qrow + c * 32 + lh * 8;
            qf[c] = mk16(*(const v8us*)p, *(const v8us*)(p + 16));
        }
    }

    float mrow[8], lrow[8];
    v8f o[4];
#pragma unroll
    for (int r = 0; r < 8; r++) { mrow[r] = -1e30f; lrow[r] = 0.0f; }
#pragma unroll
    for (int j = 0; j < 4; j++) o[j] = zero8();

    for (int ks0 = 0; ks0 < Sq; ks0 += 32) {
        // ---- S = Q * K^T  (two 16x16 score tiles) ----
        v8f sc[2];
#pragma unroll
        for (int tt = 0; tt < 2; tt++) {
            sc[tt] = zero8();
#pragma unroll
            for (int c = 0; c < 2; c++) {
                const unsigned short* kp =
                    Kw + ((size_t)(b * Sq + ks0 + tt * 16 + lm)) * Ntot + h * DK + c * 32 + lh * 8;
                v16bf kf = mk16(*(const v8us*)kp, *(const v8us*)(kp + 16));
                sc[tt] = __builtin_amdgcn_wmma_f32_16x16x32_bf16(
                    false, qf[c], false, kf, (short)0, sc[tt], false, false);
            }
        }

        // ---- online softmax (row = r + 8*lh, cols across lanes lm) ----
        float corr[8];
#pragma unroll
        for (int r = 0; r < 8; r++) {
            float s0 = sc[0][r] * 0.125f;
            float s1 = sc[1][r] * 0.125f;
            float mx = fmaxf(s0, s1);
#pragma unroll
            for (int off = 1; off < 16; off <<= 1) mx = fmaxf(mx, __shfl_xor(mx, off, 32));
            mx = fmaxf(mx, mrow[r]);
            const float c0 = __expf(mrow[r] - mx);
            const float p0 = __expf(s0 - mx);
            const float p1 = __expf(s1 - mx);
            float ps = p0 + p1;
#pragma unroll
            for (int off = 1; off < 16; off <<= 1) ps += __shfl_xor(ps, off, 32);
            lrow[r] = lrow[r] * c0 + ps;
            mrow[r] = mx;
            corr[r] = c0;
            const int row = r + 8 * lh;
            P[row * PSTR + lm]      = f2bf(p0);   // keys ks0 + 0..15
            P[row * PSTR + 16 + lm] = f2bf(p1);   // keys ks0 + 16..31
        }
#pragma unroll
        for (int j = 0; j < 4; j++)
#pragma unroll
            for (int r = 0; r < 8; r++) o[j][r] *= corr[r];

        // ---- reshuffle P (C-layout -> A-layout) via LDS; DS in-order per wave ----
        const unsigned short* pp = &P[lm * PSTR + lh * 8];
        v16bf pf = mk16(*(const v8us*)pp, *(const v8us*)(pp + 16));

        // ---- O += P * V   (V transposed: contiguous in s per (b,h,d)) ----
#pragma unroll
        for (int j = 0; j < 4; j++) {
            const unsigned short* vp =
                Vt + ((size_t)((b * NH + h) * DK + j * 16 + lm)) * Sq + ks0 + lh * 8;
            v16bf vf = mk16(*(const v8us*)vp, *(const v8us*)(vp + 16));
            o[j] = __builtin_amdgcn_wmma_f32_16x16x32_bf16(
                false, pf, false, vf, (short)0, o[j], false, false);
        }
    }

    // ---- normalize and store ctx (bf16 row-major [b,s,h*64+d]) ----
#pragma unroll
    for (int j = 0; j < 4; j++)
#pragma unroll
        for (int r = 0; r < 8; r++) {
            const int row = r + 8 * lh;
            const float val = o[j][r] / lrow[r];
            Ctx[((size_t)(b * Sq + q0 + row)) * Ntot + h * DK + j * 16 + lm] = f2bf(val);
        }
}

// =====================================================================
extern "C" void kernel_launch(void* const* d_in, const int* in_sizes, int n_in,
                              void* d_out, int out_size, void* d_ws, size_t ws_size,
                              hipStream_t stream) {
    const float* x  = (const float*)d_in[0];
    const float* wq = (const float*)d_in[1];
    const float* bq = (const float*)d_in[2];
    const float* wk = (const float*)d_in[3];
    const float* bk = (const float*)d_in[4];
    const float* wv = (const float*)d_in[5];
    const float* bv = (const float*)d_in[6];
    const float* wo = (const float*)d_in[7];
    const float* bo = (const float*)d_in[8];
    float* out = (float*)d_out;

    const size_t MAT = (size_t)Mtot * Ntot;  // elements per bf16 matrix
    unsigned short* ws  = (unsigned short*)d_ws;
    unsigned short* qws = ws;
    unsigned short* kws = ws + MAT;
    unsigned short* vws = ws + 2 * MAT;   // transposed [b,h,d,s]
    unsigned short* cws = ws + 3 * MAT;

    dim3 gg(Mtot / BM, Ntot / BN), gb(256);

    gemm_bf16_wmma<float, float, 1><<<gg, gb, 0, stream>>>(x, wq, bq, (void*)qws);
    gemm_bf16_wmma<float, float, 1><<<gg, gb, 0, stream>>>(x, wk, bk, (void*)kws);
    gemm_bf16_wmma<float, float, 2><<<gg, gb, 0, stream>>>(x, wv, bv, (void*)vws);

    attn_fwd_wmma<<<dim3(Bsz * NH * (Sq / 128)), gb, 0, stream>>>(qws, kws, vws, cws);

    gemm_bf16_wmma<unsigned short, float, 0><<<gg, gb, 0, stream>>>(cws, wo, bo, (void*)out);
}
